// BDH_39127152067244
// MI455X (gfx1250) — matrix operand order repaired and checked by reading, not compiled, not run on hardware
//
#include <hip/hip_runtime.h>

#define NLAYER  6
#define DD      256
#define NHEAD   4
#define NDIM    8192
#define TSEQ    512
#define BBATCH  2
#define BT      1024          // BBATCH*TSEQ
#define NVOCAB  256
#define TWO_PI  6.2831853071795864769f

typedef __attribute__((ext_vector_type(16))) __bf16 bf16x16;
typedef __attribute__((ext_vector_type(8)))  float  f32x8;
typedef unsigned int u32x4 __attribute__((ext_vector_type(4)));
typedef int          i32x4 __attribute__((ext_vector_type(4)));
typedef int          i32x8 __attribute__((ext_vector_type(8)));

#define LDSS 40   // LDS row stride in bf16 elems (80 B = 16 B-aligned, bank-skewed)

// ---------------------------------------------------------------- WMMA core
__device__ __forceinline__ f32x8 wmma_bf16(bf16x16 a, bf16x16 b, f32x8 c) {
  return __builtin_amdgcn_wmma_f32_16x16x32_bf16(false, a, false, b, (short)0, c,
                                                 false, false);
}

__device__ __forceinline__ bf16x16 frag_a(const __bf16* As, int wm, int lane) {
  int r  = wm * 16 + (lane & 15);
  int kh = (lane < 16) ? 0 : 8;
  const __bf16* p = As + r * LDSS + kh;
  union { uint4 u[2]; bf16x16 v; } t;
  t.u[0] = *(const uint4*)(p);
  t.u[1] = *(const uint4*)(p + 16);
  return t.v;
}

__device__ __forceinline__ bf16x16 frag_b(const __bf16* Bs, int ncol, int lane) {
  int c  = ncol + (lane & 15);
  int kh = (lane < 16) ? 0 : 8;
  const __bf16* p = Bs + c * LDSS + kh;
  union { uint4 u[2]; bf16x16 v; } t;
  t.u[0] = *(const uint4*)(p);
  t.u[1] = *(const uint4*)(p + 16);
  return t.v;
}

// ---------------------------------------------------------------- TDM staging
// 2D tile: 64 rows x 32 bf16 (64 B/row) -> LDS with 16 B pad per row (stride 80 B
// == LDSS), via Tensor Data Mover.  D# group0/group1 built per ISA 08 §8.3/8.4.
__device__ __forceinline__ void tdm2d(unsigned lds_addr, const void* gaddr,
                                      unsigned row_stride_elems) {
  unsigned long long ga = (unsigned long long)(size_t)gaddr;
  u32x4 g0;
  g0[0] = 1u;                                   // count=1, user descriptor
  g0[1] = lds_addr;                             // LDS byte address
  g0[2] = (unsigned)(ga & 0xFFFFFFFFu);         // global_addr[31:0]
  g0[3] = ((unsigned)(ga >> 32) & 0x01FFFFFFu)  // global_addr[56:32]
          | (2u << 30);                         // type=2 (image)
  i32x8 g1;
  // data_size=1 (2B) | pad_enable | pad_interval=3 (16 DW) | pad_amount=3 (4 DW)
  g1[0] = (1 << 16) | (1 << 20) | (3 << 22) | (3 << 25);
  unsigned td0 = 0x7FFFFFFFu, td1 = 0x7FFFFFFFu;   // huge dims: no OOB clipping
  g1[1] = (int)(td0 << 16);                        // [47:32]=0, td0[15:0]
  g1[2] = (int)((td0 >> 16) | (td1 << 16));        // td0[31:16], td1[15:0]
  g1[3] = (int)((td1 >> 16) | (32u << 16));        // td1[31:16], tile_dim0=32
  g1[4] = 64;                                      // tile_dim1=64, tile_dim2=0
  g1[5] = (int)row_stride_elems;                   // tensor_dim0_stride[31:0]
  g1[6] = 0;
  g1[7] = 0;
  i32x4 z4 = {0, 0, 0, 0};
#if __clang_major__ >= 23
  i32x8 z8 = {0, 0, 0, 0, 0, 0, 0, 0};
  __builtin_amdgcn_tensor_load_to_lds(g0, g1, z4, z4, z8, 0);
#else
  __builtin_amdgcn_tensor_load_to_lds(g0, g1, z4, z4, 0);
#endif
}

__device__ __forceinline__ unsigned lds_off(const void* p) {
  return (unsigned)(size_t)p;   // low 32 bits of flat LDS address = LDS offset
}

// Double-buffered TDM GEMM inner loop.  A: row-major [64 x K] (lda elems/row),
// B: pre-transposed row-major [64 x K] (ldb elems/row).  wave0 DMAs A tiles,
// wave1 DMAs B tiles, one tile in flight (wait tensorcnt<=1).
__device__ __forceinline__ void gemm_tdm(const __bf16* A, unsigned lda,
                                         const __bf16* B, unsigned ldb, int nk,
                                         __bf16* As0, __bf16* As1,
                                         __bf16* Bs0, __bf16* Bs1,
                                         int wave, int lane, int wm, int wn,
                                         f32x8& c0, f32x8& c1) {
  if (wave == 0)      tdm2d(lds_off(As0), A, lda);
  else if (wave == 1) tdm2d(lds_off(Bs0), B, ldb);
  for (int i = 0; i < nk - 1; ++i) {
    const __bf16* Ac = (i & 1) ? As1 : As0;
    const __bf16* Bc = (i & 1) ? Bs1 : Bs0;
    __bf16* An = (i & 1) ? As0 : As1;
    __bf16* Bn = (i & 1) ? Bs0 : Bs1;
    if (wave == 0)      tdm2d(lds_off(An), A + (size_t)(i + 1) * 32, lda);
    else if (wave == 1) tdm2d(lds_off(Bn), B + (size_t)(i + 1) * 32, ldb);
    __builtin_amdgcn_s_wait_tensorcnt((short)1);   // tile i landed
    __syncthreads();
    bf16x16 a  = frag_a(Ac, wm, lane);
    bf16x16 b0 = frag_b(Bc, wn * 32, lane);
    bf16x16 b1 = frag_b(Bc, wn * 32 + 16, lane);
    c0 = wmma_bf16(a, b0, c0);
    c1 = wmma_bf16(a, b1, c1);
    __syncthreads();                               // done reading buf[i&1]
  }
  const __bf16* Ac = ((nk - 1) & 1) ? As1 : As0;
  const __bf16* Bc = ((nk - 1) & 1) ? Bs1 : Bs0;
  __builtin_amdgcn_s_wait_tensorcnt((short)0);
  __syncthreads();
  bf16x16 a  = frag_a(Ac, wm, lane);
  bf16x16 b0 = frag_b(Bc, wn * 32, lane);
  bf16x16 b1 = frag_b(Bc, wn * 32 + 16, lane);
  c0 = wmma_bf16(a, b0, c0);
  c1 = wmma_bf16(a, b1, c1);
  __syncthreads();
}

// ---------------------------------------------------------------- VALU staging
// (kept for the two small f32-source GEMMs: yKV and lm_head)
__device__ __forceinline__ void stage_a_f32(__bf16* As, const float* A, int lda,
                                            int kBase) {
  int tid = threadIdx.x;
  int r   = tid >> 2;
  int seg = (tid & 3) << 3;
  const float* src = A + (size_t)r * lda + kBase + seg;
  float4 f0 = ((const float4*)src)[0];
  float4 f1 = ((const float4*)src)[1];
  union { uint4 u; __bf16 h[8]; } t;
  t.h[0]=(__bf16)f0.x; t.h[1]=(__bf16)f0.y; t.h[2]=(__bf16)f0.z; t.h[3]=(__bf16)f0.w;
  t.h[4]=(__bf16)f1.x; t.h[5]=(__bf16)f1.y; t.h[6]=(__bf16)f1.z; t.h[7]=(__bf16)f1.w;
  *(uint4*)(As + r * LDSS + seg) = t.u;
}

__device__ __forceinline__ void stage_b_rm(__bf16* Bs, const float* B, int ldb,
                                           int kBase, int nBase) {
  int tid = threadIdx.x;
  int n   = tid & 63;
  int kg  = (tid >> 6) << 3;
  const float* src = B + (size_t)(kBase + kg) * ldb + nBase + n;
  union { uint4 u; __bf16 h[8]; } t;
#pragma unroll
  for (int j = 0; j < 8; ++j) t.h[j] = (__bf16)src[(size_t)j * ldb];
  *(uint4*)(Bs + n * LDSS + kg) = t.u;
}

// ---------------------------------------------------------------- reductions
__device__ __forceinline__ float bsum256(float v, float* sb) {
  int tid = threadIdx.x;
  sb[tid] = v;
  __syncthreads();
#pragma unroll
  for (int s = 128; s >= 1; s >>= 1) {
    if (tid < s) sb[tid] += sb[tid + s];
    __syncthreads();
  }
  float r = sb[0];
  __syncthreads();
  return r;
}

// ---------------------------------------------------------------- prep kernels
__global__ void k_rope(float* __restrict__ cosT, float* __restrict__ sinT) {
  size_t p = (size_t)blockIdx.x * blockDim.x + threadIdx.x;  // pair id
  if (p >= (size_t)TSEQ * (NDIM / 2)) return;
  int t = (int)(p / (NDIM / 2));
  int n = (int)(p % (NDIM / 2)) * 2;
  float freq = exp2f(-16.0f * (float)n / (float)NDIM) * (1.0f / TWO_PI);
  float ph   = (float)t * freq;
  ph = (ph - floorf(ph)) * TWO_PI;
  float c = cosf(ph), s = sinf(ph);
  size_t o = (size_t)t * NDIM + n;
  cosT[o] = c; cosT[o + 1] = c;
  sinT[o] = s; sinT[o + 1] = s;
}

// 32x32 tiled transpose + f32->bf16: dst[C x R] = src[R x C]^T  (per z slice)
__global__ void k_tcvt(const float* __restrict__ src, __bf16* __restrict__ dst,
                       int R, int C) {
  __shared__ float tile[32][33];
  int c0 = blockIdx.x * 32, r0 = blockIdx.y * 32;
  size_t slice = (size_t)blockIdx.z * R * C;
  int tx = threadIdx.x & 31, ty = threadIdx.x >> 5;
#pragma unroll
  for (int j = 0; j < 32; j += 8)
    tile[ty + j][tx] = src[slice + (size_t)(r0 + ty + j) * C + c0 + tx];
  __syncthreads();
#pragma unroll
  for (int j = 0; j < 32; j += 8)
    dst[slice + (size_t)(c0 + ty + j) * R + r0 + tx] = (__bf16)tile[tx][ty + j];
}

__global__ void k_embed(const int* __restrict__ idx, const float* __restrict__ ew,
                        float* __restrict__ x, __bf16* __restrict__ x_bf) {
  __shared__ float sb[256];
  int row = blockIdx.x, tid = threadIdx.x;
  float v = ew[(size_t)idx[row] * DD + tid];
  float m = bsum256(v, sb) * (1.0f / DD);
  float d = v - m;
  float var = bsum256(d * d, sb) * (1.0f / DD);
  float r = d * rsqrtf(var + 1e-5f);
  x[(size_t)row * DD + tid]    = r;
  x_bf[(size_t)row * DD + tid] = (__bf16)r;
}

// ---------------------------------------------------------------- layer kernels
// x_sparse = relu(x @ enc_h); fused RoPE -> qr.  TDM staged.  M=1024 N=8192 K=256
__global__ void k_enc(const __bf16* __restrict__ x_bf, const __bf16* __restrict__ encT,
                      const float* __restrict__ cosT, const float* __restrict__ sinT,
                      __bf16* __restrict__ xs_bf, __bf16* __restrict__ qr_bf) {
  __shared__ __align__(16) __bf16 As0[64 * LDSS], As1[64 * LDSS];
  __shared__ __align__(16) __bf16 Bs0[64 * LDSS], Bs1[64 * LDSS];
  int nBase = blockIdx.x * 64, mBase = blockIdx.y * 64, h = blockIdx.z;
  const __bf16* A = x_bf + (size_t)mBase * DD;
  const __bf16* B = encT + (size_t)h * NDIM * DD + (size_t)nBase * DD;
  int lane = threadIdx.x & 31, wave = threadIdx.x >> 5;
  int wm = wave & 3, wn = wave >> 2;
  f32x8 c0 = {}, c1 = {};
  gemm_tdm(A, DD, B, DD, DD / 32, As0, As1, Bs0, Bs1, wave, lane, wm, wn, c0, c1);
  int rl0 = wm * 16 + ((lane < 16) ? 0 : 8);
  int col = lane & 15;
  float sgn = (lane & 1) ? 1.f : -1.f;   // even n: v*c - p*s ; odd n: v*c + p*s
#pragma unroll
  for (int e = 0; e < 8; ++e) {
    int m = mBase + rl0 + e;
    int b = m >> 9, t = m & 511;
    size_t ro = ((size_t)(b * NHEAD + h) * TSEQ + t) * NDIM;
    int n0 = nBase + wn * 32 + col;
    float v0 = c0[e]; v0 = v0 > 0.f ? v0 : 0.f;
    float v1 = c1[e]; v1 = v1 > 0.f ? v1 : 0.f;
    float p0 = __shfl_xor(v0, 1, 32);
    float p1 = __shfl_xor(v1, 1, 32);
    size_t to = (size_t)t * NDIM + n0;
    float cc0 = cosT[to],      ss0 = sinT[to];
    float cc1 = cosT[to + 16], ss1 = sinT[to + 16];
    xs_bf[ro + n0]      = (__bf16)v0;
    xs_bf[ro + n0 + 16] = (__bf16)v1;
    qr_bf[ro + n0]      = (__bf16)(v0 * cc0 + sgn * p0 * ss0);
    qr_bf[ro + n0 + 16] = (__bf16)(v1 * cc1 + sgn * p1 * ss1);
  }
}

// scores = qr @ qr^T with strict-lower mask.  TDM staged.  K=8192
__global__ void k_scores(const __bf16* __restrict__ qr_bf, float* __restrict__ scores) {
  int sBase = blockIdx.x * 64, tBase = blockIdx.y * 64, bh = blockIdx.z;
  float* out = scores + (size_t)bh * TSEQ * TSEQ;
  if (sBase >= tBase + 64) {                 // fully masked tile
    for (int i = threadIdx.x; i < 64 * 64; i += 256) {
      int r = i >> 6, c = i & 63;
      out[(size_t)(tBase + r) * TSEQ + sBase + c] = 0.f;
    }
    return;
  }
  __shared__ __align__(16) __bf16 As0[64 * LDSS], As1[64 * LDSS];
  __shared__ __align__(16) __bf16 Bs0[64 * LDSS], Bs1[64 * LDSS];
  const __bf16* qbh = qr_bf + (size_t)bh * TSEQ * NDIM;
  const __bf16* A = qbh + (size_t)tBase * NDIM;
  const __bf16* B = qbh + (size_t)sBase * NDIM;
  int lane = threadIdx.x & 31, wave = threadIdx.x >> 5;
  int wm = wave & 3, wn = wave >> 2;
  f32x8 c0 = {}, c1 = {};
  gemm_tdm(A, NDIM, B, NDIM, NDIM / 32, As0, As1, Bs0, Bs1, wave, lane, wm, wn, c0, c1);
  int rl0 = wm * 16 + ((lane < 16) ? 0 : 8);
  int col = lane & 15;
#pragma unroll
  for (int e = 0; e < 8; ++e) {
    int t  = tBase + rl0 + e;
    int s0 = sBase + wn * 32 + col;
    out[(size_t)t * TSEQ + s0]      = (s0 < t)      ? c0[e] : 0.f;
    out[(size_t)t * TSEQ + s0 + 16] = (s0 + 16 < t) ? c1[e] : 0.f;
  }
}

// yKV = scores @ x_b   (f32 sources, VALU staging)  M=512 N=256 K=512
__global__ void k_yKV(const float* __restrict__ scores, const float* __restrict__ x,
                      float* __restrict__ yKV) {
  __shared__ __align__(16) __bf16 As[64 * LDSS];
  __shared__ __align__(16) __bf16 Bs[64 * LDSS];
  int nBase = blockIdx.x * 64, mBase = blockIdx.y * 64, bh = blockIdx.z;
  int b = bh >> 2;
  const float* A = scores + (size_t)bh * TSEQ * TSEQ + (size_t)mBase * TSEQ;
  const float* B = x + (size_t)b * TSEQ * DD;
  int lane = threadIdx.x & 31, wave = threadIdx.x >> 5;
  int wm = wave & 3, wn = wave >> 2;
  f32x8 c0 = {}, c1 = {};
  for (int kB = 0; kB < TSEQ; kB += 32) {
    __syncthreads();
    stage_a_f32(As, A, TSEQ, kB);
    stage_b_rm(Bs, B, DD, kB, nBase);
    __syncthreads();
    bf16x16 a  = frag_a(As, wm, lane);
    bf16x16 b0 = frag_b(Bs, wn * 32, lane);
    bf16x16 b1 = frag_b(Bs, wn * 32 + 16, lane);
    c0 = wmma_bf16(a, b0, c0);
    c1 = wmma_bf16(a, b1, c1);
  }
  int rl0 = wm * 16 + ((lane < 16) ? 0 : 8);
  int col = lane & 15;
#pragma unroll
  for (int e = 0; e < 8; ++e) {
    size_t ro = ((size_t)bh * TSEQ + (mBase + rl0 + e)) * DD;
    int d0 = nBase + wn * 32 + col;
    yKV[ro + d0]      = c0[e];
    yKV[ro + d0 + 16] = c1[e];
  }
}

// LN over last dim 256; f32 in -> bf16 out
__global__ void k_ln(const float* __restrict__ src, __bf16* __restrict__ dst) {
  __shared__ float sb[256];
  int row = blockIdx.x, tid = threadIdx.x;
  float v = src[(size_t)row * DD + tid];
  float m = bsum256(v, sb) * (1.0f / DD);
  float d = v - m;
  float var = bsum256(d * d, sb) * (1.0f / DD);
  dst[(size_t)row * DD + tid] = (__bf16)(d * rsqrtf(var + 1e-5f));
}

// xy = relu(yKV_ln @ encv_h) * xs.  TDM staged.  M=512 N=8192 K=256 per bh
__global__ void k_gate(const __bf16* __restrict__ yKV_bf, const __bf16* __restrict__ encvT,
                       const __bf16* __restrict__ xs_bf, __bf16* __restrict__ xy_bf) {
  __shared__ __align__(16) __bf16 As0[64 * LDSS], As1[64 * LDSS];
  __shared__ __align__(16) __bf16 Bs0[64 * LDSS], Bs1[64 * LDSS];
  int nBase = blockIdx.x * 64, mBase = blockIdx.y * 64, bh = blockIdx.z;
  int h = bh & 3;
  const __bf16* A = yKV_bf + (size_t)bh * TSEQ * DD + (size_t)mBase * DD;
  const __bf16* B = encvT + (size_t)h * NDIM * DD + (size_t)nBase * DD;
  int lane = threadIdx.x & 31, wave = threadIdx.x >> 5;
  int wm = wave & 3, wn = wave >> 2;
  f32x8 c0 = {}, c1 = {};
  gemm_tdm(A, DD, B, DD, DD / 32, As0, As1, Bs0, Bs1, wave, lane, wm, wn, c0, c1);
  int rl0 = wm * 16 + ((lane < 16) ? 0 : 8);
  int col = lane & 15;
#pragma unroll
  for (int e = 0; e < 8; ++e) {
    size_t ro = ((size_t)bh * TSEQ + (mBase + rl0 + e)) * NDIM;
    int n0 = nBase + wn * 32 + col;
    float v0 = c0[e]; v0 = v0 > 0.f ? v0 : 0.f;
    float v1 = c1[e]; v1 = v1 > 0.f ? v1 : 0.f;
    xy_bf[ro + n0]      = (__bf16)(v0 * (float)xs_bf[ro + n0]);
    xy_bf[ro + n0 + 16] = (__bf16)(v1 * (float)xs_bf[ro + n0 + 16]);
  }
}

// yMLP = sum_h xy_h @ dec_h.  TDM staged.  M=1024 N=256 K=4*8192
__global__ void k_dec(const __bf16* __restrict__ xy_bf, const __bf16* __restrict__ decT,
                      float* __restrict__ yMLP) {
  __shared__ __align__(16) __bf16 As0[64 * LDSS], As1[64 * LDSS];
  __shared__ __align__(16) __bf16 Bs0[64 * LDSS], Bs1[64 * LDSS];
  int nBase = blockIdx.x * 64, mBase = blockIdx.y * 64;
  int b = mBase >> 9, tBase = mBase & 511;
  int lane = threadIdx.x & 31, wave = threadIdx.x >> 5;
  int wm = wave & 3, wn = wave >> 2;
  f32x8 c0 = {}, c1 = {};
  for (int h = 0; h < NHEAD; ++h) {
    const __bf16* A = xy_bf + ((size_t)(b * NHEAD + h) * TSEQ + tBase) * NDIM;
    const __bf16* B = decT + (size_t)h * DD * NDIM + (size_t)nBase * NDIM;
    gemm_tdm(A, NDIM, B, NDIM, NDIM / 32, As0, As1, Bs0, Bs1, wave, lane, wm, wn,
             c0, c1);
  }
  int rl0 = wm * 16 + ((lane < 16) ? 0 : 8);
  int col = lane & 15;
#pragma unroll
  for (int e = 0; e < 8; ++e) {
    size_t ro = (size_t)(mBase + rl0 + e) * DD;
    int d0 = nBase + wn * 32 + col;
    yMLP[ro + d0]      = c0[e];
    yMLP[ro + d0 + 16] = c1[e];
  }
}

__global__ void k_resln(float* __restrict__ x, __bf16* __restrict__ x_bf,
                        const float* __restrict__ yMLP) {
  __shared__ float sb[256];
  int row = blockIdx.x, tid = threadIdx.x;
  float y = yMLP[(size_t)row * DD + tid];
  float m = bsum256(y, sb) * (1.0f / DD);
  float d = y - m;
  float var = bsum256(d * d, sb) * (1.0f / DD);
  float v = x[(size_t)row * DD + tid] + d * rsqrtf(var + 1e-5f);
  float m2 = bsum256(v, sb) * (1.0f / DD);
  float d2 = v - m2;
  float v2 = bsum256(d2 * d2, sb) * (1.0f / DD);
  float r = d2 * rsqrtf(v2 + 1e-5f);
  x[(size_t)row * DD + tid]    = r;
  x_bf[(size_t)row * DD + tid] = (__bf16)r;
}

// logits = x @ lm_head  (f32 sources, VALU staging)  M=1024 N=256 K=256
__global__ void k_head(const float* __restrict__ x, const float* __restrict__ lmh,
                       float* __restrict__ out) {
  __shared__ __align__(16) __bf16 As[64 * LDSS];
  __shared__ __align__(16) __bf16 Bs[64 * LDSS];
  int nBase = blockIdx.x * 64, mBase = blockIdx.y * 64;
  const float* A = x + (size_t)mBase * DD;
  int lane = threadIdx.x & 31, wave = threadIdx.x >> 5;
  int wm = wave & 3, wn = wave >> 2;
  f32x8 c0 = {}, c1 = {};
  for (int kB = 0; kB < DD; kB += 32) {
    __syncthreads();
    stage_a_f32(As, A, DD, kB);
    stage_b_rm(Bs, lmh, NVOCAB, kB, nBase);
    __syncthreads();
    bf16x16 a  = frag_a(As, wm, lane);
    bf16x16 b0 = frag_b(Bs, wn * 32, lane);
    bf16x16 b1 = frag_b(Bs, wn * 32 + 16, lane);
    c0 = wmma_bf16(a, b0, c0);
    c1 = wmma_bf16(a, b1, c1);
  }
  int rl0 = wm * 16 + ((lane < 16) ? 0 : 8);
  int col = lane & 15;
#pragma unroll
  for (int e = 0; e < 8; ++e) {
    size_t ro = (size_t)(mBase + rl0 + e) * NVOCAB;
    int d0 = nBase + wn * 32 + col;
    out[ro + d0]      = c0[e];
    out[ro + d0 + 16] = c1[e];
  }
}

// ---------------------------------------------------------------- host
extern "C" void kernel_launch(void* const* d_in, const int* in_sizes, int n_in,
                              void* d_out, int out_size, void* d_ws, size_t ws_size,
                              hipStream_t stream) {
  (void)in_sizes; (void)n_in; (void)out_size; (void)ws_size;
  const int*   idx  = (const int*)  d_in[0];
  const float* ew   = (const float*)d_in[1];
  const float* enc  = (const float*)d_in[2];
  const float* encv = (const float*)d_in[3];
  const float* dec  = (const float*)d_in[4];
  const float* lmh  = (const float*)d_in[5];
  float* out = (float*)d_out;

  const size_t TN  = (size_t)TSEQ * NDIM;
  const size_t BIG = (size_t)BBATCH * NHEAD * TN;
  char* p = (char*)d_ws;
  auto alloc = [&](size_t bytes) {
    char* r = p;
    p += (bytes + 255) & ~(size_t)255;
    return r;
  };
  float*  cosT   = (float*) alloc(TN * 4);
  float*  sinT   = (float*) alloc(TN * 4);
  __bf16* xs_bf  = (__bf16*)alloc(BIG * 2);
  __bf16* qr_bf  = (__bf16*)alloc(BIG * 2);
  __bf16* xy_bf  = (__bf16*)alloc(BIG * 2);
  float*  scores = (float*) alloc((size_t)BBATCH * NHEAD * TSEQ * TSEQ * 4);
  float*  yKV    = (float*) alloc((size_t)BBATCH * NHEAD * TSEQ * DD * 4);
  __bf16* yKV_bf = (__bf16*)alloc((size_t)BBATCH * NHEAD * TSEQ * DD * 2);
  float*  yMLP   = (float*) alloc((size_t)BT * DD * 4);
  float*  x      = (float*) alloc((size_t)BT * DD * 4);
  __bf16* x_bf   = (__bf16*)alloc((size_t)BT * DD * 2);
  __bf16* encT   = (__bf16*)alloc((size_t)NHEAD * NDIM * DD * 2);
  __bf16* encvT  = (__bf16*)alloc((size_t)NHEAD * NDIM * DD * 2);
  __bf16* decT   = (__bf16*)alloc((size_t)NHEAD * NDIM * DD * 2);

  dim3 blk(256);
  k_rope<<<(TSEQ * (NDIM / 2)) / 256, blk, 0, stream>>>(cosT, sinT);
  // weights: f32 -> bf16, transposed to [n'][k] layout for TDM B-tiles
  k_tcvt<<<dim3(NDIM / 32, DD / 32, NHEAD), blk, 0, stream>>>(enc, encT, DD, NDIM);
  k_tcvt<<<dim3(NDIM / 32, DD / 32, NHEAD), blk, 0, stream>>>(encv, encvT, DD, NDIM);
  k_tcvt<<<dim3(DD / 32, NDIM / 32, NHEAD), blk, 0, stream>>>(dec, decT, NDIM, DD);
  k_embed<<<BT, blk, 0, stream>>>(idx, ew, x, x_bf);
  for (int l = 0; l < NLAYER; ++l) {
    k_enc<<<dim3(NDIM / 64, BT / 64, NHEAD), blk, 0, stream>>>(x_bf, encT, cosT, sinT,
                                                               xs_bf, qr_bf);
    k_scores<<<dim3(TSEQ / 64, TSEQ / 64, BBATCH * NHEAD), blk, 0, stream>>>(qr_bf,
                                                                             scores);
    k_yKV<<<dim3(DD / 64, TSEQ / 64, BBATCH * NHEAD), blk, 0, stream>>>(scores, x, yKV);
    k_ln<<<BBATCH * NHEAD * TSEQ, blk, 0, stream>>>(yKV, yKV_bf);
    k_gate<<<dim3(NDIM / 64, TSEQ / 64, BBATCH * NHEAD), blk, 0, stream>>>(
        yKV_bf, encvT, xs_bf, xy_bf);
    k_dec<<<dim3(DD / 64, BT / 64), blk, 0, stream>>>(xy_bf, decT, yMLP);
    k_resln<<<BT, blk, 0, stream>>>(x, x_bf, yMLP);
  }
  k_head<<<dim3(NVOCAB / 64, BT / 64), blk, 0, stream>>>(x, lmh, out);
}
